// ChebNet_15444702396429
// MI455X (gfx1250) — matrix operand order repaired and checked by reading, not compiled, not run on hardware
//
#include <hip/hip_runtime.h>

// ============================================================================
// ChebNet (K=39/43/45/1, C=2->14->20->27->1) on N=100K nodes, E=3.2M edges.
// Memory-bound edge scatter (entire working set ~65MB is L2-resident on the
// MI455X's 192MB L2) + WMMA f16 16x16x32 for the dense combine steps, with
// the Chebyshev recurrence and bias-init fused into the WMMA kernel.
// ============================================================================

typedef __attribute__((ext_vector_type(16))) _Float16 v16h;
typedef __attribute__((ext_vector_type(8)))  float    v8f;

#define CP 32   // padded feature stride == WMMA K dimension

__device__ __forceinline__ void atomicAddF(float* p, float v) {
  // lowers to global_atomic_add_f32 (no-return -> STOREcnt path)
  (void)__hip_atomic_fetch_add(p, v, __ATOMIC_RELAXED, __HIP_MEMORY_SCOPE_AGENT);
}

// deg[src] += w  (weighted out-degree, matches segment_sum over src)
__global__ void deg_kernel(const int* __restrict__ src, const float* __restrict__ w,
                           float* __restrict__ deg, int E) {
  int e = blockIdx.x * blockDim.x + threadIdx.x;
  if (e < E) atomicAddF(&deg[src[e]], w[e]);
}

// norm_e = -d_src^-1/2 * w_e * d_dst^-1/2   (inf -> 0)
__global__ void norm_kernel(const int* __restrict__ src, const int* __restrict__ dst,
                            const float* __restrict__ w, const float* __restrict__ deg,
                            float* __restrict__ nrm, int E) {
  int e = blockIdx.x * blockDim.x + threadIdx.x;
  if (e >= E) return;
  float ds = deg[src[e]], dd = deg[dst[e]];
  float is = ds > 0.f ? rsqrtf(ds) : 0.f;
  float id = dd > 0.f ? rsqrtf(dd) : 0.f;
  nrm[e] = -is * w[e] * id;
}

// h[n*32+c] = x[n*Cin+c] for c<Cin else 0  (zero padding feeds WMMA K=32)
__global__ void pad_kernel(const float* __restrict__ x, float* __restrict__ h,
                           int N, int Cin) {
  int i = blockIdx.x * blockDim.x + threadIdx.x;
  if (i >= N * CP) return;
  int c = i & (CP - 1), n = i >> 5;
  h[i] = (c < Cin) ? x[n * Cin + c] : 0.f;
}

// hout[dst] += norm_e * hin[src], C == 2 fast path (float2 gather)
__global__ void prop2_kernel(const int* __restrict__ src, const int* __restrict__ dst,
                             const float* __restrict__ nrm,
                             const float* __restrict__ hin, float* __restrict__ hout,
                             int E) {
  int e = blockIdx.x * blockDim.x + threadIdx.x;
  if (e >= E) return;
  float nv = nrm[e];
  float2 v = *(const float2*)(hin + (size_t)src[e] * CP);
  float* hd = hout + (size_t)dst[e] * CP;
  atomicAddF(hd + 0, nv * v.x);
  atomicAddF(hd + 1, nv * v.y);
}

// hout[dst] += norm_e * hin[src] over C4 float4 chunks (pads are zero, so the
// rounded-up chunks only add 0.0 to zero pad columns -> harmless).
__global__ void prop4_kernel(const int* __restrict__ src, const int* __restrict__ dst,
                             const float* __restrict__ nrm,
                             const float* __restrict__ hin, float* __restrict__ hout,
                             int E, int C4) {
  int e = blockIdx.x * blockDim.x + threadIdx.x;
  if (e >= E) return;
  float nv = nrm[e];
  const float4* hs = (const float4*)(hin + (size_t)src[e] * CP);
  float* hd = hout + (size_t)dst[e] * CP;
#pragma unroll 2
  for (int q = 0; q < C4; ++q) {
    float4 v = hs[q];                 // global_load_b128 gather
    atomicAddF(hd + 4 * q + 0, nv * v.x);
    atomicAddF(hd + 4 * q + 1, nv * v.y);
    atomicAddF(hd + 4 * q + 2, nv * v.z);
    atomicAddF(hd + 4 * q + 3, nv * v.w);
  }
}

// SiLU(acc) -> out   (silu(0)=0, pads stay zero)
__global__ void silu_kernel(const float* __restrict__ acc, float* __restrict__ out,
                            int n) {
  int i = blockIdx.x * blockDim.x + threadIdx.x;
  if (i >= n) return;
  float v = acc[i];
  out[i] = v / (1.f + __expf(-v));
}

// ----------------------------------------------------------------------------
// Fused combine step:  acc(16x16 tile) += A @ W[Cin x Cout]
//   do_cheb: A = 2*X - Xm2 (Chebyshev recurrence), ct==0 blocks write A back
//            to X (each element covered exactly once per co-tile -> race-free)
//   do_init: C fragment initialized from bias (k=0) instead of loading acc
// One V_WMMA_F32_16X16X32_F16 per wave; X stride CP=32 with zero pads means a
// single WMMA covers the full contraction. One wave per block (EXEC all-ones).
// N must be a multiple of 16 (100000 = 6250*16). Grid.y is always 2 so all 32
// acc columns get written (pad co-tile computes 0 via the W/bias guards).
// ----------------------------------------------------------------------------
__global__ void wmma_mm_kernel(float* __restrict__ X, const float* __restrict__ Xm2,
                               const float* __restrict__ W,
                               const float* __restrict__ bias,
                               float* __restrict__ acc, int Cin, int Cout,
                               int do_cheb, int do_init) {
  const int lane = threadIdx.x & 31;
  const int hf = lane >> 4;        // lane half: 0 = lanes 0-15, 1 = lanes 16-31
  const int ln = lane & 15;
  const int nt = blockIdx.x;       // 16-node tile
  const int ct = blockIdx.y;       // 16-outcol tile

  // ---- A fragment: per-lane cols are two contiguous 8-float runs ->
  //      four float4 (global_load_b128) loads.
  const size_t rowoff = ((size_t)nt * 16 + ln) * CP;
  float* xr = X + rowoff;
  const int c0 = 8 * hf;           // cols c0..c0+7   -> a[0..7]
  const int c1 = 16 + 8 * hf;      // cols c1..c1+7   -> a[8..15]
  float4 x0 = *(const float4*)(xr + c0);
  float4 x1 = *(const float4*)(xr + c0 + 4);
  float4 x2 = *(const float4*)(xr + c1);
  float4 x3 = *(const float4*)(xr + c1 + 4);

  if (do_cheb) {                   // tx = 2*prop - tx_m2, fused into A packing
    const float* mr = Xm2 + rowoff;
    float4 m0 = *(const float4*)(mr + c0);
    float4 m1 = *(const float4*)(mr + c0 + 4);
    float4 m2 = *(const float4*)(mr + c1);
    float4 m3 = *(const float4*)(mr + c1 + 4);
    x0.x = 2.f*x0.x - m0.x; x0.y = 2.f*x0.y - m0.y; x0.z = 2.f*x0.z - m0.z; x0.w = 2.f*x0.w - m0.w;
    x1.x = 2.f*x1.x - m1.x; x1.y = 2.f*x1.y - m1.y; x1.z = 2.f*x1.z - m1.z; x1.w = 2.f*x1.w - m1.w;
    x2.x = 2.f*x2.x - m2.x; x2.y = 2.f*x2.y - m2.y; x2.z = 2.f*x2.z - m2.z; x2.w = 2.f*x2.w - m2.w;
    x3.x = 2.f*x3.x - m3.x; x3.y = 2.f*x3.y - m3.y; x3.z = 2.f*x3.z - m3.z; x3.w = 2.f*x3.w - m3.w;
    if (ct == 0) {                 // store tx back (becomes tx_m1 next step)
      *(float4*)(xr + c0)     = x0;
      *(float4*)(xr + c0 + 4) = x1;
      *(float4*)(xr + c1)     = x2;
      *(float4*)(xr + c1 + 4) = x3;
    }
  }

  v16h a;
  a[0]=(_Float16)x0.x; a[1]=(_Float16)x0.y; a[2]=(_Float16)x0.z; a[3]=(_Float16)x0.w;
  a[4]=(_Float16)x1.x; a[5]=(_Float16)x1.y; a[6]=(_Float16)x1.z; a[7]=(_Float16)x1.w;
  a[8]=(_Float16)x2.x; a[9]=(_Float16)x2.y; a[10]=(_Float16)x2.z; a[11]=(_Float16)x2.w;
  a[12]=(_Float16)x3.x; a[13]=(_Float16)x3.y; a[14]=(_Float16)x3.z; a[15]=(_Float16)x3.w;

  // ---- B fragment: 32x16 f16, b[s]: K=8*hf+s (s<8), K=16+8*hf+(s-8) (s>=8).
  const int co = ct * 16 + ln;
  v16h b;
#pragma unroll
  for (int v = 0; v < 8; ++v) {
    int kb = 2 * v + 8 * hf + 8 * (v >> 2);
    float w0 = (kb     < Cin && co < Cout) ? W[(size_t)kb * Cout + co]       : 0.f;
    float w1 = (kb + 1 < Cin && co < Cout) ? W[(size_t)(kb + 1) * Cout + co] : 0.f;
    b[2 * v]     = (_Float16)w0;
    b[2 * v + 1] = (_Float16)w1;
  }

  // ---- C/D: f32 16x16, VGPR r: lanes0-15 M=r, lanes16-31 M=8+r.
  float* ac = acc + ((size_t)nt * 16 + hf * 8) * CP + (size_t)ct * 16 + ln;
  v8f c;
  if (do_init) {
    float bv = (co < Cout) ? bias[co] : 0.f;
#pragma unroll
    for (int r = 0; r < 8; ++r) c[r] = bv;
  } else {
#pragma unroll
    for (int r = 0; r < 8; ++r) c[r] = ac[(size_t)r * CP];
  }

  c = __builtin_amdgcn_wmma_f32_16x16x32_f16(false, a, false, b, (short)0, c,
                                             false, false);

#pragma unroll
  for (int r = 0; r < 8; ++r) ac[(size_t)r * CP] = c[r];
}

// out[n] = sigmoid( dot(h[n,0:27], W4) )   (K=1, no bias)
__global__ void final_kernel(const float* __restrict__ H, const float* __restrict__ w,
                             float* __restrict__ out, int N, int Cin) {
  int n = blockIdx.x * blockDim.x + threadIdx.x;
  if (n >= N) return;
  const float* h = H + (size_t)n * CP;
  float s = 0.f;
#pragma unroll 9
  for (int c = 0; c < Cin; ++c) s += h[c] * w[c];
  out[n] = 1.f / (1.f + __expf(-s));
}

static inline int ceil_div(int a, int b) { return (a + b - 1) / b; }

extern "C" void kernel_launch(void* const* d_in, const int* in_sizes, int n_in,
                              void* d_out, int out_size, void* d_ws, size_t ws_size,
                              hipStream_t stream) {
  const float* x  = (const float*)d_in[0];   // N x 2
  const int*   ei = (const int*)d_in[1];     // 2 x E (integer inputs -> int32)
  const float* ew = (const float*)d_in[2];   // E
  const float* Wl[4] = {(const float*)d_in[3], (const float*)d_in[5],
                        (const float*)d_in[7], (const float*)d_in[9]};
  const float* bl[3] = {(const float*)d_in[4], (const float*)d_in[6],
                        (const float*)d_in[8]};
  const int Kl[3]     = {39, 43, 45};
  const int Cin_l[3]  = {2, 14, 20};
  const int Cout_l[3] = {14, 20, 27};

  const int N = in_sizes[0] / 2;
  const int E = in_sizes[2];
  const int* src = ei;
  const int* dst = ei + E;

  // ---- workspace carve (~65 MB; L2-resident working set on MI455X) ----
  char* wsb = (char*)d_ws;
  size_t off = 0;
  auto take = [&](size_t bytes) -> char* {
    char* p = wsb + off;
    off = (off + bytes + 255) & ~(size_t)255;
    return p;
  };
  float* deg = (float*)take((size_t)N * 4);
  float* nrm = (float*)take((size_t)E * 4);
  float* F[4];
  for (int i = 0; i < 4; ++i) F[i] = (float*)take((size_t)N * CP * 4);
  (void)ws_size;

  const int TB = 256;
  const int egrid = ceil_div(E, TB);
  const int ngrid = ceil_div(N * CP, TB);
  const dim3 g(N / 16, 2);   // all 32 acc columns covered every call

  // ---- graph normalization (computed once, reused by all 124 propagations)
  hipMemsetAsync(deg, 0, (size_t)N * 4, stream);
  deg_kernel<<<egrid, TB, 0, stream>>>(src, ew, deg, E);
  norm_kernel<<<egrid, TB, 0, stream>>>(src, dst, ew, deg, nrm, E);
  pad_kernel<<<ngrid, TB, 0, stream>>>(x, F[0], N, 2);

  float* hin = F[0];
  float* acc = F[1];
  float* t1  = F[2];
  float* t2  = F[3];

  for (int L = 0; L < 3; ++L) {
    const int K = Kl[L], Ci = Cin_l[L], Co = Cout_l[L];
    const int C4 = (Ci + 3) / 4;

    // k = 0: acc = bias + x @ W[0]   (bias init fused into WMMA)
    wmma_mm_kernel<<<g, 32, 0, stream>>>(hin, nullptr, Wl[L], bl[L], acc,
                                         Ci, Co, 0, 1);

    // k = 1: tx1 = prop(x); acc += tx1 @ W[1]
    float *pm2 = hin, *pm1 = t1, *pf = t2;
    hipMemsetAsync(pm1, 0, (size_t)N * CP * 4, stream);
    if (Ci == 2) prop2_kernel<<<egrid, TB, 0, stream>>>(src, dst, nrm, hin, pm1, E);
    else         prop4_kernel<<<egrid, TB, 0, stream>>>(src, dst, nrm, hin, pm1, E, C4);
    wmma_mm_kernel<<<g, 32, 0, stream>>>(pm1, nullptr, Wl[L] + (size_t)Ci * Co,
                                         nullptr, acc, Ci, Co, 0, 0);

    // k >= 2: tx = 2*prop(tx_m1) - tx_m2 (fused in WMMA); acc += tx @ W[k]
    for (int k = 2; k < K; ++k) {
      hipMemsetAsync(pf, 0, (size_t)N * CP * 4, stream);
      if (Ci == 2) prop2_kernel<<<egrid, TB, 0, stream>>>(src, dst, nrm, pm1, pf, E);
      else         prop4_kernel<<<egrid, TB, 0, stream>>>(src, dst, nrm, pm1, pf, E, C4);
      wmma_mm_kernel<<<g, 32, 0, stream>>>(pf, pm2, Wl[L] + (size_t)k * Ci * Co,
                                           nullptr, acc, Ci, Co, 1, 0);
      float* tmp = pm2; pm2 = pm1; pm1 = pf; pf = tmp;  // rotate 3 buffers
    }

    // SiLU into the now-free rotation buffer -> next layer input
    silu_kernel<<<ngrid, TB, 0, stream>>>(acc, pf, N * CP);
    hin = pf; t1 = pm2; t2 = pm1;  // acc reused as-is next layer
  }

  // final layer: K=1, 27 -> 1, sigmoid
  final_kernel<<<ceil_div(N, TB), TB, 0, stream>>>(hin, Wl[3], (float*)d_out, N, 27);
}
